// ContextModelBase_36043365548512
// MI455X (gfx1250) — compile-verified, matrix operand order checked by reading, not run on hardware
//
#include <hip/hip_runtime.h>

typedef __attribute__((ext_vector_type(16))) _Float16 v16h;
typedef __attribute__((ext_vector_type(8)))  float    v8f;

#define C_IN    128
#define C_OUT   256
#define KEFF    62            // only first 62 of 125 taps have non-zero weights
#define NK      125
#define NPTS    200000
#define MTILE   32
#define LDS_PAD 8
#define LDA     (C_IN + LDS_PAD)    // 136 halfs: stride 272B -> conflict-free A reads
#define LDACT   (C_OUT + LDS_PAD)   // 264 halfs: stride 528B -> conflict-free A reads

union HFrag {
    v16h h;
    unsigned int u[8];
    uint4 q[2];
};

static __device__ inline unsigned int pack2(float a, float b) {
    _Float16 ha = (_Float16)a, hb = (_Float16)b;
    unsigned short ua = __builtin_bit_cast(unsigned short, ha);
    unsigned short ub = __builtin_bit_cast(unsigned short, hb);
    return (unsigned int)ua | ((unsigned int)ub << 16);
}

static __device__ inline v8f wmma16(const HFrag& a, const HFrag& b, v8f c) {
    return __builtin_amdgcn_wmma_f32_16x16x32_f16(
        false, a.h, false, b.h, (short)0, c, false, false);
}

// ---------------------------------------------------------------------------
// One-time f32 -> f16 conversion of feats, plus an explicit zero row at index
// NPTS (== feats_pad), so the gather loop has no bounds branch and EXEC stays
// all-ones around the WMMA pipeline.
// ---------------------------------------------------------------------------
__global__ void convert_feats_kernel(const float* __restrict__ f,
                                     unsigned int* __restrict__ fh) {
    const long total = (long)NPTS * C_IN / 4;         // float4 groups
    long t = (long)blockIdx.x * blockDim.x + threadIdx.x;
    if (t < total) {
        const float4 v = ((const float4*)f)[t];
        uint2 o; o.x = pack2(v.x, v.y); o.y = pack2(v.z, v.w);
        ((uint2*)fh)[t] = o;
    } else if (t < total + C_IN / 4) {
        uint2 z; z.x = 0u; z.y = 0u;
        ((uint2*)fh)[t] = z;                          // zero pad row
    }
}

// ---------------------------------------------------------------------------
// Pack a (Cin x Cout) f32 matrix (K-dim = rows) into the f16 B-fragment layout
// for v_wmma_f32_16x16x32_f16:
//   packed[mat][ct][kc][lane][j] (uint, 2 halfs):
//     n  = ct*16 + (lane&15)
//     k0 = kc*32 + (j>=4?16:0) + (lane>>4)*8 + (j&3)*2
// A wave then fetches one 32x16 fragment as two coalesced global_load_b128.
// ---------------------------------------------------------------------------
__global__ void pack_b_kernel(const float* __restrict__ W, unsigned int* __restrict__ P,
                              int Cin, int Cout, int nmat, int matStride) {
    const int numKc = Cin / 32;
    const long total = (long)nmat * (Cout / 16) * numKc * 32 * 8;
    long tid = (long)blockIdx.x * blockDim.x + threadIdx.x;
    if (tid >= total) return;

    int j  = (int)(tid & 7);
    int L  = (int)((tid >> 3) & 31);
    long r = tid >> 8;
    int kc = (int)(r % numKc); r /= numKc;
    int ct = (int)(r % (Cout / 16)); r /= (Cout / 16);
    int m  = (int)r;

    int n  = ct * 16 + (L & 15);
    int k0 = kc * 32 + ((j >= 4) ? 16 : 0) + (L >> 4) * 8 + (j & 3) * 2;

    const float* Wb = W + (long)m * matStride;
    P[tid] = pack2(Wb[(long)k0 * Cout + n], Wb[(long)(k0 + 1) * Cout + n]);
}

// ---------------------------------------------------------------------------
// Fused gather-conv (62 taps) + 3-layer MLP.
// Block = 32 points, 256 threads = 8 wave32s. Each wave owns a 2x2 fragment
// tile: both 16-row tiles x 2 of the 16 column tiles (4 accumulators).
// Gather staging uses CDNA5 async loads straight into LDS (ASYNCcnt).
// ---------------------------------------------------------------------------
__global__ void __launch_bounds__(256)
ctx_mlp_kernel(const _Float16* __restrict__ featsH,
               const int*      __restrict__ neigh,
               const unsigned int* __restrict__ WmP,
               const float* __restrict__ bm,
               const unsigned int* __restrict__ W0P, const float* __restrict__ b0,
               const unsigned int* __restrict__ W1P, const float* __restrict__ b1,
               const unsigned int* __restrict__ W2P, const float* __restrict__ b2,
               float* __restrict__ out) {
    __shared__ __align__(16) _Float16 stage[2][MTILE][LDA];   // double-buffered gather
    __shared__ __align__(16) _Float16 act[MTILE][LDACT];      // inter-layer activations

    const int tid  = threadIdx.x;
    const int lane = tid & 31;
    const int wave = tid >> 5;
    const int n0   = blockIdx.x * MTILE;
    const int ct0  = wave * 2;
    const int ct1  = wave * 2 + 1;

    const int grow = tid >> 4;          // 0..15 (row within half-tile)
    const int gcol = (tid & 15) * 8;    // 8-half slab

    // Async gather: one global_load_async_to_lds_b128 per half-tile per wave.
    // idx is always valid (row NPTS of featsH is the zero pad row).
    auto gather_async = [&](int k, int buf) {
#pragma unroll
        for (int h = 0; h < 2; ++h) {
            const int row = grow + h * 16;
            const int idx = __builtin_nontemporal_load(&neigh[(long)(n0 + row) * NK + k]);
            unsigned long long src =
                (unsigned long long)(const void*)(featsH + (long)idx * C_IN + gcol);
            unsigned int dst =
                (unsigned int)(unsigned long long)(const void*)&stage[buf][row][gcol];
            asm volatile("global_load_async_to_lds_b128 %0, %1, off"
                         :: "v"(dst), "v"(src) : "memory");
        }
    };

    // A fragment (16x32 f16) from padded row-major LDS buffer
    auto load_a = [&](const _Float16* base, int ld, int rt, int kcBase, HFrag& a) {
        const unsigned int* u = (const unsigned int*)base;
        const int m  = (lane & 15) + rt * 16;
        const int h  = lane >> 4;
        const int bi = (m * ld + kcBase + h * 8) >> 1;
        a.u[0] = u[bi + 0]; a.u[1] = u[bi + 1]; a.u[2] = u[bi + 2];  a.u[3] = u[bi + 3];
        a.u[4] = u[bi + 8]; a.u[5] = u[bi + 9]; a.u[6] = u[bi + 10]; a.u[7] = u[bi + 11];
    };

    // B fragment (32x16 f16): two coalesced b128 from packed weights (L2-resident)
    auto load_b = [&](const unsigned int* P, int ct, int kc, int numKc, HFrag& b) {
        const uint4* q = (const uint4*)(P + (((long)(ct * numKc + kc) * 32 + lane) * 8));
        b.q[0] = q[0];
        b.q[1] = q[1];
    };

    // ---------------- Phase 1: gather conv over 62 taps ----------------
    v8f a00 = {}, a01 = {}, a10 = {}, a11 = {};   // [row tile][col tile]
    gather_async(0, 0);
    for (int k = 0; k < KEFF; ++k) {
        asm volatile("s_wait_asynccnt 0x0" ::: "memory");
        __syncthreads();
        if (k + 1 < KEFF) gather_async(k + 1, (k + 1) & 1);
        const _Float16* A = &stage[k & 1][0][0];
        const unsigned int* Pk = WmP + (long)k * (16 * 4 * 32 * 8);
#pragma unroll
        for (int kc = 0; kc < 4; ++kc) {
            HFrag fa0, fa1, fb0, fb1;
            load_a(A, LDA, 0, kc * 32, fa0);
            load_a(A, LDA, 1, kc * 32, fa1);
            load_b(Pk, ct0, kc, 4, fb0);
            load_b(Pk, ct1, kc, 4, fb1);
            a00 = wmma16(fa0, fb0, a00);
            a01 = wmma16(fa0, fb1, a01);
            a10 = wmma16(fa1, fb0, a10);
            a11 = wmma16(fa1, fb1, a11);
        }
    }

    // add bm bias (accumulator layout: col = lane&15, rows in the 8 VGPRs)
    {
        const float bv0 = bm[ct0 * 16 + (lane & 15)];
        const float bv1 = bm[ct1 * 16 + (lane & 15)];
#pragma unroll
        for (int r = 0; r < 8; ++r) {
            a00[r] += bv0; a01[r] += bv1;
            a10[r] += bv0; a11[r] += bv1;
        }
    }

    auto store_act = [&](const v8f& a, int rt, int ct) {
        const int n  = ct * 16 + (lane & 15);
        const int mh = rt * 16 + (lane >> 4) * 8;
#pragma unroll
        for (int r = 0; r < 8; ++r) act[mh + r][n] = (_Float16)a[r];
    };

    auto layer = [&](const unsigned int* WP, const float* bias, bool doRelu,
                     v8f& o00, v8f& o01, v8f& o10, v8f& o11) {
        o00 = (v8f){}; o01 = (v8f){}; o10 = (v8f){}; o11 = (v8f){};
#pragma unroll
        for (int kc = 0; kc < 8; ++kc) {
            HFrag fa0, fa1, fb0, fb1;
            load_a(&act[0][0], LDACT, 0, kc * 32, fa0);
            load_a(&act[0][0], LDACT, 1, kc * 32, fa1);
            load_b(WP, ct0, kc, 8, fb0);
            load_b(WP, ct1, kc, 8, fb1);
            o00 = wmma16(fa0, fb0, o00);
            o01 = wmma16(fa0, fb1, o01);
            o10 = wmma16(fa1, fb0, o10);
            o11 = wmma16(fa1, fb1, o11);
        }
        const float bv0 = bias[ct0 * 16 + (lane & 15)];
        const float bv1 = bias[ct1 * 16 + (lane & 15)];
#pragma unroll
        for (int r = 0; r < 8; ++r) {
            o00[r] += bv0; o01[r] += bv1; o10[r] += bv0; o11[r] += bv1;
            if (doRelu) {
                o00[r] = fmaxf(o00[r], 0.f); o01[r] = fmaxf(o01[r], 0.f);
                o10[r] = fmaxf(o10[r], 0.f); o11[r] = fmaxf(o11[r], 0.f);
            }
        }
    };

    // ---------------- Phase 2: fused MLP ----------------
    store_act(a00, 0, ct0); store_act(a01, 0, ct1);
    store_act(a10, 1, ct0); store_act(a11, 1, ct1);
    __syncthreads();
    v8f h00, h01, h10, h11;
    layer(W0P, b0, true, h00, h01, h10, h11);
    __syncthreads();
    store_act(h00, 0, ct0); store_act(h01, 0, ct1);
    store_act(h10, 1, ct0); store_act(h11, 1, ct1);
    __syncthreads();
    layer(W1P, b1, true, h00, h01, h10, h11);
    __syncthreads();
    store_act(h00, 0, ct0); store_act(h01, 0, ct1);
    store_act(h10, 1, ct0); store_act(h11, 1, ct1);
    __syncthreads();
    layer(W2P, b2, false, h00, h01, h10, h11);

    // ------- Output: loc = out[:, :128], scale = |out[:, 128:]| (NT stores) ---
    auto write_out = [&](const v8f& o, int rt, int ct) {
        const int col = ct * 16 + (lane & 15);
        const int mh  = rt * 16 + (lane >> 4) * 8;
        const bool isScale = (col >= 128);
        const int c = isScale ? (col - 128) : col;
        float* dst = out + (isScale ? (long)NPTS * 128 : 0);
#pragma unroll
        for (int r = 0; r < 8; ++r) {
            float v = o[r];
            if (isScale) v = fabsf(v);
            __builtin_nontemporal_store(v, &dst[(long)(n0 + mh + r) * 128 + c]);
        }
    };
    write_out(h00, 0, ct0); write_out(h01, 0, ct1);
    write_out(h10, 1, ct0); write_out(h11, 1, ct1);
}

// ---------------------------------------------------------------------------
extern "C" void kernel_launch(void* const* d_in, const int* in_sizes, int n_in,
                              void* d_out, int out_size, void* d_ws, size_t ws_size,
                              hipStream_t stream) {
    const float* feats = (const float*)d_in[0];
    const int*   neigh = (const int*)d_in[1];
    const float* Wm    = (const float*)d_in[2];
    const float* bm    = (const float*)d_in[3];
    const float* W0    = (const float*)d_in[4];
    const float* b0    = (const float*)d_in[5];
    const float* W1    = (const float*)d_in[6];
    const float* b1    = (const float*)d_in[7];
    const float* W2    = (const float*)d_in[8];
    const float* b2    = (const float*)d_in[9];
    float* out = (float*)d_out;

    // workspace layout:
    //   featsH : (NPTS+1) x 128 f16  = 51,200,256 B (multiple of 256)
    //   WmP    : 62 * 64 KB packed f16 fragments
    //   W0P/W1P/W2P : 128 KB each
    _Float16*     featsH = (_Float16*)d_ws;
    unsigned int* WmP = (unsigned int*)((char*)d_ws + (size_t)(NPTS + 1) * C_IN * 2);
    unsigned int* W0P = WmP + (size_t)KEFF * 16384;
    unsigned int* W1P = W0P + 32768;
    unsigned int* W2P = W1P + 32768;

    {
        long total = (long)NPTS * C_IN / 4 + C_IN / 4;
        convert_feats_kernel<<<(int)((total + 255) / 256), 256, 0, stream>>>(
            feats, (unsigned int*)featsH);
    }
    {
        long total = (long)KEFF * 16384;
        pack_b_kernel<<<(int)((total + 255) / 256), 256, 0, stream>>>(
            Wm, WmP, C_IN, C_OUT, KEFF, C_IN * C_OUT);
    }
    pack_b_kernel<<<32768 / 256, 256, 0, stream>>>(W0, W0P, C_OUT, C_OUT, 1, 0);
    pack_b_kernel<<<32768 / 256, 256, 0, stream>>>(W1, W1P, C_OUT, C_OUT, 1, 0);
    pack_b_kernel<<<32768 / 256, 256, 0, stream>>>(W2, W2P, C_OUT, C_OUT, 1, 0);

    ctx_mlp_kernel<<<NPTS / MTILE, 256, 0, stream>>>(
        featsH, neigh, WmP, bm, W0P, b0, W1P, b1, W2P, b2, out);
}
